// Encoder_Decoder_MultiHead_Attention_85392539779895
// MI455X (gfx1250) — matrix operand order repaired and checked by reading, not compiled, not run on hardware
//
#include <hip/hip_runtime.h>

#define BB  4
#define SS  2048
#define EE  512
#define HH  8
#define DHH 64

typedef __attribute__((ext_vector_type(16))) __bf16 v16bf;
typedef __attribute__((ext_vector_type(8)))  float  v8f;

static __device__ __forceinline__ v8f wmma_bf16(v16bf a, v16bf b, v8f c) {
  return __builtin_amdgcn_wmma_f32_16x16x32_bf16(false, a, false, b, (short)0, c, false, false);
}

// A fragment (16x32 bf16, M x K): lane holds row (lane&15); halves of K per ISA layout.
// p points at &A[row][k_tile_base + kbA], contiguous K.
static __device__ __forceinline__ v16bf load_a_f32(const float* __restrict__ p) {
  v16bf a;
#pragma unroll
  for (int j = 0; j < 8; ++j) { a[j] = (__bf16)p[j]; a[8 + j] = (__bf16)p[16 + j]; }
  return a;
}

// B fragment (32x16 bf16, K x N): lane holds col (lane&15); same K split as A:
// p points at K base+8*half; elements 0..7 = p[0..7], 8..15 = p[16..23].
static __device__ __forceinline__ v16bf load_b_bf16_contig(const __bf16* __restrict__ p) {
  v16bf b;
#pragma unroll
  for (int j = 0; j < 16; ++j) b[j] = p[j < 8 ? j : j + 8];
  return b;
}

// ---------------------------------------------------------------------------
// Kernel 0: pre-pack weights into bf16 once per call.
//   WQ/WK/WV : [H,E,DH] f32  ->  [H,DH,E] bf16  (transposed: contiguous along E)
//   WO       : [E,E]    f32  ->  [E,E]    bf16
// Note H*DH*E == E*E == 262144, so one grid covers all four tensors.
// ---------------------------------------------------------------------------
__global__ __launch_bounds__(256)
void mha_pack_weights(const float* __restrict__ WQ, const float* __restrict__ WK,
                      const float* __restrict__ WV, const float* __restrict__ WO,
                      __bf16* __restrict__ WQt, __bf16* __restrict__ WKt,
                      __bf16* __restrict__ WVt, __bf16* __restrict__ WOb) {
  const int idx = blockIdx.x * 256 + threadIdx.x;   // 0 .. 262143
  const int h   = idx / (DHH * EE);
  const int rem = idx % (DHH * EE);
  const int d   = rem / EE;
  const int e   = rem % EE;
  const size_t src = ((size_t)h * EE + e) * DHH + d;
  WQt[idx] = (__bf16)WQ[src];
  WKt[idx] = (__bf16)WK[src];
  WVt[idx] = (__bf16)WV[src];
  WOb[idx] = (__bf16)WO[idx];
}

// ---------------------------------------------------------------------------
// Kernel 1: per-head Q/K/V projections. Output Q,K: [B,H,S,DH] bf16; V stored
// transposed [B,H,DH,S] bf16 so PV B-fragments are contiguous later.
// Block: 128 threads = 4 waves; wave w owns DH columns [16w, 16w+16); block owns 16 S rows.
// ---------------------------------------------------------------------------
__global__ __launch_bounds__(128)
void mha_qkv_proj(const float* __restrict__ eq, const float* __restrict__ ek,
                  const float* __restrict__ ev,
                  const __bf16* __restrict__ WQt, const float* __restrict__ bQ,
                  const __bf16* __restrict__ WKt, const float* __restrict__ bK,
                  const __bf16* __restrict__ WVt, const float* __restrict__ bV,
                  __bf16* __restrict__ Qh, __bf16* __restrict__ Kh, __bf16* __restrict__ Vt) {
  const int lane = threadIdx.x & 31;
  const int wave = threadIdx.x >> 5;     // N tile of DH
  const int half = lane >> 4;
  const int l15  = lane & 15;
  const int b = blockIdx.z, h = blockIdx.y;
  const int mbase = blockIdx.x * 16;
  const int kbA = half * 8;
  const int kbB = half * 16;
  const int d   = wave * 16 + l15;
  const int rowA = mbase + l15;

  const size_t arow = ((size_t)b * SS + rowA) * EE;
  const float* pq = eq + arow;
  const float* pk = ek + arow;
  const float* pv = ev + arow;
  // packed weights: row d of [DH,E], contiguous along E (the K dim)
  const __bf16* wq = WQt + ((size_t)h * DHH + d) * EE + kbA;
  const __bf16* wk = WKt + ((size_t)h * DHH + d) * EE + kbA;
  const __bf16* wv = WVt + ((size_t)h * DHH + d) * EE + kbA;

  v8f cq = {}, ck = {}, cv = {};
  for (int kk = 0; kk < EE; kk += 32) {
    v16bf aq = load_a_f32(pq + kk + kbA);
    v16bf bq = load_b_bf16_contig(wq + kk);
    cq = wmma_bf16(aq, bq, cq);
    v16bf ak = load_a_f32(pk + kk + kbA);
    v16bf bk = load_b_bf16_contig(wk + kk);
    ck = wmma_bf16(ak, bk, ck);
    v16bf av = load_a_f32(pv + kk + kbA);
    v16bf bv = load_b_bf16_contig(wv + kk);
    cv = wmma_bf16(av, bv, cv);
  }

  const float biasQ = bQ[h * DHH + d];
  const float biasK = bK[h * DHH + d];
  const float biasV = bV[h * DHH + d];
  const size_t head = (size_t)(b * HH + h);
#pragma unroll
  for (int r = 0; r < 8; ++r) {
    const int m = mbase + r + half * 8;            // global S row per C layout
    const size_t idx = (head * SS + m) * DHH + d;
    Qh[idx] = (__bf16)(cq[r] + biasQ);
    Kh[idx] = (__bf16)(ck[r] + biasK);
    Vt[(head * DHH + d) * SS + m] = (__bf16)(cv[r] + biasV);
  }
}

// ---------------------------------------------------------------------------
// Kernel 2: fused flash attention with double-buffered K/V staging.
// Block = 256 threads (8 waves), 128 query rows; key loop in tiles of 32 with
// online softmax (exp2 domain). One __syncthreads per key tile.
// ---------------------------------------------------------------------------
__global__ __launch_bounds__(256)
void mha_flash_attn(const __bf16* __restrict__ Qh, const __bf16* __restrict__ Kh,
                    const __bf16* __restrict__ Vt, __bf16* __restrict__ cat) {
  __shared__ __bf16 ldsK[2][32][DHH];    // [buf][key][dh]   2 x 4 KB
  __shared__ __bf16 ldsV[2][DHH][32];    // [buf][dh][key]   2 x 4 KB
  __shared__ __bf16 ldsP[8][16][32];     // per-wave P tile, 8 KB

  const int tid  = threadIdx.x;
  const int lane = tid & 31;
  const int wave = tid >> 5;
  const int half = lane >> 4;
  const int l15  = lane & 15;
  const int b = blockIdx.z, h = blockIdx.y;
  const int qbase = blockIdx.x * 128 + wave * 16;
  const size_t head = (size_t)(b * HH + h);
  const size_t qkbase = head * SS * DHH;
  const int kbA = half * 8;
  const int kbB = half * 16;

  // Q fragments for dh 0..31 and 32..63
  v16bf aq0, aq1;
  {
    const __bf16* qrow = Qh + qkbase + (size_t)(qbase + l15) * DHH;
#pragma unroll
    for (int j = 0; j < 8; ++j) {
      aq0[j] = qrow[kbA + j];        aq0[8 + j] = qrow[kbA + 16 + j];
      aq1[j] = qrow[32 + kbA + j];   aq1[8 + j] = qrow[32 + kbA + 16 + j];
    }
  }

  float mrow[8], lrow[8];
#pragma unroll
  for (int r = 0; r < 8; ++r) { mrow[r] = -1e30f; lrow[r] = 0.0f; }
  v8f o[4] = {v8f{}, v8f{}, v8f{}, v8f{}};

  // softmax in exp2 domain: fold 1/sqrt(DH) * log2(e) into one scale
  const float scale2 = 0.125f * 1.44269504088896f;

  // staging coords: 16 bytes / thread / tile
  const int krow = tid >> 3, kcol = (tid & 7) * 8;   // K tile: 32 x 64
  const int vrow = tid >> 2, vcol = (tid & 3) * 8;   // Vt tile: 64 x 32

  // prologue: stage tile 0 into buffer 0
  *reinterpret_cast<int4*>(&ldsK[0][krow][kcol]) =
      *reinterpret_cast<const int4*>(Kh + qkbase + (size_t)krow * DHH + kcol);
  *reinterpret_cast<int4*>(&ldsV[0][vrow][vcol]) =
      *reinterpret_cast<const int4*>(Vt + (head * DHH + vrow) * SS + vcol);
  __syncthreads();

  for (int kt = 0; kt < SS; kt += 32) {
    const int buf = (kt >> 5) & 1;
    const bool more = (kt + 32) < SS;

    // issue next tile's global loads now; commit to LDS after the P fence
    int4 kreg = {}, vreg = {};
    if (more) {
      const __bf16* gK = Kh + qkbase + (size_t)(kt + 32 + krow) * DHH + kcol;
      const __bf16* gV = Vt + (head * DHH + vrow) * SS + (kt + 32) + vcol;
      kreg = *reinterpret_cast<const int4*>(gK);
      vreg = *reinterpret_cast<const int4*>(gV);
      if (kt + 64 < SS) {
        // prefetch distance-2 tiles (lowers to global_prefetch_b8)
        __builtin_prefetch(gK + (size_t)32 * DHH, 0, 1);
        __builtin_prefetch(gV + 32, 0, 1);
      }
    }

    // scores: c0 = keys kt..kt+15, c1 = keys kt+16..kt+31  (K-dim = DH = 64 -> 2 WMMAs each)
    v8f c0 = {}, c1 = {};
    {
      const __bf16* kr0 = &ldsK[buf][l15][0];
      const __bf16* kr1 = &ldsK[buf][16 + l15][0];
      v16bf bk;
#pragma unroll
      for (int j = 0; j < 16; ++j) bk[j] = kr0[kbA + (j < 8 ? j : j + 8)];
      c0 = wmma_bf16(aq0, bk, c0);
#pragma unroll
      for (int j = 0; j < 16; ++j) bk[j] = kr0[32 + kbA + (j < 8 ? j : j + 8)];
      c0 = wmma_bf16(aq1, bk, c0);
#pragma unroll
      for (int j = 0; j < 16; ++j) bk[j] = kr1[kbA + (j < 8 ? j : j + 8)];
      c1 = wmma_bf16(aq0, bk, c1);
#pragma unroll
      for (int j = 0; j < 16; ++j) bk[j] = kr1[32 + kbA + (j < 8 ? j : j + 8)];
      c1 = wmma_bf16(aq1, bk, c1);
    }

    // online softmax per row (each row lives in 16 lanes of one half-wave)
#pragma unroll
    for (int r = 0; r < 8; ++r) {
      float x0 = c0[r] * scale2;
      float x1 = c1[r] * scale2;
      float mx = fmaxf(x0, x1);
#pragma unroll
      for (int off = 1; off < 16; off <<= 1) mx = fmaxf(mx, __shfl_xor(mx, off, 32));
      float mnew = fmaxf(mrow[r], mx);
      float corr = exp2f(mrow[r] - mnew);
      float e0 = exp2f(x0 - mnew);
      float e1 = exp2f(x1 - mnew);
      float rs = e0 + e1;
#pragma unroll
      for (int off = 1; off < 16; off <<= 1) rs += __shfl_xor(rs, off, 32);
      lrow[r] = lrow[r] * corr + rs;
      mrow[r] = mnew;
      o[0][r] *= corr; o[1][r] *= corr; o[2][r] *= corr; o[3][r] *= corr;
      const int m = r + half * 8;
      ldsP[wave][m][l15]      = (__bf16)e0;
      ldsP[wave][m][16 + l15] = (__bf16)e1;
    }

    // wave-local fence: P region is private per wave, order ds stores vs loads
    __builtin_amdgcn_wave_barrier();
    asm volatile("s_wait_dscnt 0" ::: "memory");
    __builtin_amdgcn_wave_barrier();

    // re-layout P (C layout) into an A fragment
    v16bf ap;
    {
      const __bf16* pr = &ldsP[wave][l15][0];
#pragma unroll
      for (int j = 0; j < 8; ++j) { ap[j] = pr[kbA + j]; ap[8 + j] = pr[kbA + 16 + j]; }
    }

    // commit next tile into the alternate buffer (overlaps with PV WMMAs)
    if (more) {
      *reinterpret_cast<int4*>(&ldsK[buf ^ 1][krow][kcol]) = kreg;
      *reinterpret_cast<int4*>(&ldsV[buf ^ 1][vrow][vcol]) = vreg;
    }

    // O += P x V : 4 dh tiles, K-dim = 32 keys
#pragma unroll
    for (int t = 0; t < 4; ++t) {
      v16bf bv;
      const __bf16* vr = &ldsV[buf][t * 16 + l15][0];
#pragma unroll
      for (int j = 0; j < 16; ++j) bv[j] = vr[kbA + (j < 8 ? j : j + 8)];
      o[t] = wmma_bf16(ap, bv, o[t]);
    }

    // single barrier: publishes buf^1 for next iter, retires all reads of buf
    __syncthreads();
  }

  // normalize and write head-concatenated context [B,S,E]
#pragma unroll
  for (int r = 0; r < 8; ++r) {
    const float inv = 1.0f / lrow[r];
    const int s = qbase + r + half * 8;
#pragma unroll
    for (int t = 0; t < 4; ++t) {
      const int col = h * DHH + t * 16 + l15;
      cat[((size_t)b * SS + s) * EE + col] = (__bf16)(o[t][r] * inv);
    }
  }
}

// ---------------------------------------------------------------------------
// Kernel 3: output projection  out = cat @ WO^T + bO, f32 output.
// Block = 256 threads (8 waves) covering 32 rows x 64 cols.
// ---------------------------------------------------------------------------
__global__ __launch_bounds__(256)
void mha_out_proj(const __bf16* __restrict__ cat, const __bf16* __restrict__ WOb,
                  const float* __restrict__ bO, float* __restrict__ out) {
  const int lane = threadIdx.x & 31;
  const int wave = threadIdx.x >> 5;
  const int half = lane >> 4;
  const int l15  = lane & 15;
  const int mt = wave >> 2, nt = wave & 3;
  const int mbase = blockIdx.x * 32 + mt * 16;
  const int nbase = blockIdx.y * 64 + nt * 16;
  const int kbA = half * 8, kbB = half * 16;
  const int rowA = mbase + l15;
  const int n = nbase + l15;

  const __bf16* arow = cat + (size_t)rowA * EE;
  const __bf16* brow = WOb + (size_t)n * EE + kbA;  // B[k][n] = WO[n][k], contiguous in k
  v8f acc = {};
  for (int kk = 0; kk < EE; kk += 32) {
    v16bf a;
#pragma unroll
    for (int j = 0; j < 8; ++j) { a[j] = arow[kk + kbA + j]; a[8 + j] = arow[kk + kbA + 16 + j]; }
    v16bf bfr = load_b_bf16_contig(brow + kk);
    acc = wmma_bf16(a, bfr, acc);
  }
  const float bias = bO[n];
#pragma unroll
  for (int r = 0; r < 8; ++r) {
    const int m = mbase + r + half * 8;
    out[(size_t)m * EE + n] = acc[r] + bias;
  }
}

// ---------------------------------------------------------------------------
extern "C" void kernel_launch(void* const* d_in, const int* in_sizes, int n_in,
                              void* d_out, int out_size, void* d_ws, size_t ws_size,
                              hipStream_t stream) {
  const float* eq = (const float*)d_in[0];
  const float* ek = (const float*)d_in[1];
  const float* ev = (const float*)d_in[2];
  const float* WQ = (const float*)d_in[3];
  const float* bQ = (const float*)d_in[4];
  const float* WK = (const float*)d_in[5];
  const float* bK = (const float*)d_in[6];
  const float* WV = (const float*)d_in[7];
  const float* bV = (const float*)d_in[8];
  const float* WO = (const float*)d_in[9];
  const float* bO = (const float*)d_in[10];

  const size_t perTensor = (size_t)BB * HH * SS * DHH;   // 4 Mi elements
  const size_t perWeight = (size_t)HH * DHH * EE;        // 256 Ki elements (== E*E)
  __bf16* Qh  = (__bf16*)d_ws;
  __bf16* Kh  = Qh + perTensor;
  __bf16* Vt  = Kh + perTensor;                          // transposed V [B,H,DH,S]
  __bf16* cat = Vt + perTensor;                          // [B,S,E]
  __bf16* WQt = cat + perTensor;                         // [H,DH,E]
  __bf16* WKt = WQt + perWeight;
  __bf16* WVt = WKt + perWeight;
  __bf16* WOb = WVt + perWeight;                         // [E,E]

  mha_pack_weights<<<perWeight / 256, 256, 0, stream>>>(WQ, WK, WV, WO,
                                                        WQt, WKt, WVt, WOb);
  mha_qkv_proj<<<dim3(SS / 16, HH, BB), 128, 0, stream>>>(
      eq, ek, ev, WQt, bQ, WKt, bK, WVt, bV, Qh, Kh, Vt);
  mha_flash_attn<<<dim3(SS / 128, HH, BB), 256, 0, stream>>>(Qh, Kh, Vt, cat);
  mha_out_proj<<<dim3((BB * SS) / 32, EE / 64), 256, 0, stream>>>(
      cat, WOb, bO, (float*)d_out);
}
